// LSTMNetwork_70188355551604
// MI455X (gfx1250) — compile-verified, matrix operand order checked by reading, not run on hardware
//
#include <hip/hip_runtime.h>
#include <hip/hip_bf16.h>

// Problem constants (match reference)
#define B_ 128
#define T_ 256
#define E_ 512
#define H_ 1024
#define FH_ 4096   // 4*H

typedef __attribute__((ext_vector_type(16))) __bf16 v16bf;
typedef __attribute__((ext_vector_type(8)))  float  v8f;
typedef unsigned int u32x4 __attribute__((ext_vector_type(4)));
typedef int          i32x8 __attribute__((ext_vector_type(8)));
typedef int          i32x4 __attribute__((ext_vector_type(4)));

union FragU { v16bf v; unsigned int u[8]; };

#define WMMA_BF16(a, b, c) \
  __builtin_amdgcn_wmma_f32_16x16x32_bf16(false, (a), false, (b), (short)0, (c), false, false)

#if defined(__HIP_DEVICE_COMPILE__) && __has_builtin(__builtin_amdgcn_tensor_load_to_lds) && \
    __has_builtin(__builtin_amdgcn_s_wait_tensorcnt)
#define CDNA5_HAS_TDM 1
#warning "CDNA5 probe: TDM tensor_load_to_lds path ENABLED"
#else
#define CDNA5_HAS_TDM 0
#if defined(__HIP_DEVICE_COMPILE__)
#warning "CDNA5 probe: TDM builtin absent -> cooperative-copy fallback"
#endif
#endif

__device__ __forceinline__ unsigned short f32_to_bf16_rne(float f) {
    unsigned int u = __float_as_uint(f);
    u += 0x7FFFu + ((u >> 16) & 1u);
    return (unsigned short)(u >> 16);
}

__device__ __forceinline__ float sigmoidf_(float x) {
    return 1.0f / (1.0f + __expf(-x));
}

// bf16 WMMA fragment (A row-major / B as rows of W = cols of W^T).
// ISA 7.12.2 (16-bit A 16x32): lanes 0-15 cover K {0..7,16..23}, lanes 16-31 {8..15,24..31}.
// Two contiguous 16-byte chunks -> compiler emits b128 pairs (global or ds).
__device__ __forceinline__ v16bf load_frag_bf16(const unsigned short* __restrict__ base,
                                                int row, int ld, int k0, int khalf) {
    FragU f;
    const unsigned short* p = base + (size_t)row * ld + k0 + khalf * 8;
    f.u[0] = *(const unsigned int*)(p + 0);
    f.u[1] = *(const unsigned int*)(p + 2);
    f.u[2] = *(const unsigned int*)(p + 4);
    f.u[3] = *(const unsigned int*)(p + 6);
    f.u[4] = *(const unsigned int*)(p + 16);
    f.u[5] = *(const unsigned int*)(p + 18);
    f.u[6] = *(const unsigned int*)(p + 20);
    f.u[7] = *(const unsigned int*)(p + 22);
    return f.v;
}

// Stage a [32 x H_] bf16 row-tile from global into LDS (64 KB).
// Preferred path: Tensor Data Mover (D# per ISA §8.3-8.4), issued by wave 0 only.
__device__ __forceinline__ void stage_tile(unsigned short* __restrict__ s_a,
                                           const unsigned short* __restrict__ g_rows) {
#if CDNA5_HAS_TDM
    if (threadIdx.y == 0) {
        unsigned long long ga = (unsigned long long)(size_t)g_rows;
        unsigned int lds_addr = (unsigned int)(size_t)s_a;  // low 32 bits = LDS offset
        u32x4 g0;
        g0[0] = 1u;                                          // count=1, user mode
        g0[1] = lds_addr;                                    // lds_addr
        g0[2] = (unsigned int)ga;                            // global_addr[31:0]
        g0[3] = ((unsigned int)(ga >> 32) & 0x01FFFFFFu)     // global_addr[56:32]
                | (2u << 30);                                // type=2 (image)
        i32x8 g1;
        g1[0] = (int)(1u << 16);                 // workgroup_mask=0, data_size=1 (2 bytes)
        g1[1] = (int)((H_ & 0xFFFFu) << 16);     // tensor_dim0[15:0]=1024
        g1[2] = (int)(32u << 16);                // tensor_dim0[31:16]=0 | tensor_dim1[15:0]=32
        g1[3] = (int)((unsigned)H_ << 16);       // tensor_dim1[31:16]=0 | tile_dim0=1024
        g1[4] = (int)32u;                        // tile_dim1=32, tile_dim2=0
        g1[5] = (int)(unsigned)H_;               // tensor_dim0_stride[31:0]=1024
        g1[6] = 0;                               // stride0 hi | dim1_stride lo
        g1[7] = 0;
        i32x4 gz = {0, 0, 0, 0};
#if __clang_major__ >= 23
        i32x8 gz8 = {0, 0, 0, 0, 0, 0, 0, 0};
        __builtin_amdgcn_tensor_load_to_lds(g0, g1, gz, gz, gz8, 0);
#else
        __builtin_amdgcn_tensor_load_to_lds(g0, g1, gz, gz, 0);
#endif
        __builtin_amdgcn_s_wait_tensorcnt(0);
    }
#else
    // cooperative fallback: 256 threads copy 64 KB as b128
    const uint4* s = (const uint4*)g_rows;
    uint4* d = (uint4*)s_a;
    for (int i = threadIdx.y * 32 + threadIdx.x; i < (32 * H_) / 8; i += 256) d[i] = s[i];
#endif
    __syncthreads();
}

// ---------------- prepass kernels ----------------

__global__ void cvt_f32_bf16_k(const float* __restrict__ src,
                               unsigned short* __restrict__ dst, int n) {
    int i = blockIdx.x * blockDim.x + threadIdx.x;
    if (i < n) dst[i] = f32_to_bf16_rne(src[i]);
}

__global__ void combine_bias_k(const float* __restrict__ bih0, const float* __restrict__ bhh0,
                               const float* __restrict__ bihr, const float* __restrict__ bhhr,
                               float* __restrict__ bias) {
    int i = blockIdx.x * blockDim.x + threadIdx.x;   // 0..4095
    if (i < FH_) {
        bias[i]           = bih0[i] + bhh0[i];
        bias[FH_ + i]     = bihr[i] + bhhr[i];
        bias[2 * FH_ + i] = bihr[FH_ + i] + bhhr[FH_ + i];
    }
}

__global__ void zero_u32_k(unsigned int* __restrict__ p, int n) {
    int i = blockIdx.x * blockDim.x + threadIdx.x;
    if (i < n) p[i] = 0u;
}

// ---------------- layer 0: time-parallel GEMM + activation ----------------
// Layer-0 state is always zero (faithful reference bug) -> h0[t] = f(x_t), f-gate dead.
__global__ __launch_bounds__(256) void lstm_layer0_wmma(
    const unsigned short* __restrict__ xb,   // [B*T, E] bf16 (row = b*T + t)
    const unsigned short* __restrict__ w,    // [4H, E]  bf16
    const float* __restrict__ bias,          // [4H] (b_ih0 + b_hh0)
    unsigned short* __restrict__ h0)         // [T*B, H] bf16
{
    const int wave  = blockIdx.x * 8 + threadIdx.y;   // 2048*64 wave-tiles
    const int mtile = (wave >> 6) << 4;               // M in [0, 32768)
    const int ntile = (wave & 63) << 4;               // n in [0, 1024)
    const int lane  = threadIdx.x;
    const int khalf = lane >> 4;
    const int lm    = lane & 15;
    const int arow  = mtile + lm;
    const int nr    = ntile + lm;

    v8f ai = {}, ag = {}, ao = {};
    for (int k = 0; k < E_; k += 32) {
        v16bf a  = load_frag_bf16(xb, arow, E_, k, khalf);
        v16bf bi = load_frag_bf16(w, nr,          E_, k, khalf);
        v16bf bg = load_frag_bf16(w, nr + 2 * H_, E_, k, khalf);
        v16bf bo = load_frag_bf16(w, nr + 3 * H_, E_, k, khalf);
        ai = WMMA_BF16(a, bi, ai);
        ag = WMMA_BF16(a, bg, ag);
        ao = WMMA_BF16(a, bo, ao);
    }
    const float bi_ = bias[nr];
    const float bg_ = bias[nr + 2 * H_];
    const float bo_ = bias[nr + 3 * H_];
#pragma unroll
    for (int v = 0; v < 8; ++v) {
        const int m = mtile + khalf * 8 + v;          // row of x = b*T + t
        const int bb = m >> 8;                        // / T_
        const int tt = m & (T_ - 1);                  // % T_
        float gi = ai[v] + bi_;
        float gg = ag[v] + bg_;
        float go = ao[v] + bo_;
        float c  = sigmoidf_(gi) * tanhf(gg);
        float h  = sigmoidf_(go) * tanhf(c);
        h0[((size_t)tt * B_ + bb) * H_ + nr] = f32_to_bf16_rne(h);
    }
}

// ---------------- recurrent LSTM cell (one layer, one timestep) ----------------
// Block = 8 waves, all sharing one 32-row M-pair; A rows staged to LDS (TDM),
// each wave computes 2 M-tiles x 4 gate-tiles (8 WMMAs / K-step, B frags reused).
__global__ __launch_bounds__(256) void lstm_cell_wmma(
    const unsigned short* __restrict__ hin,    // [B, H] bf16  (input from layer below)
    const unsigned short* __restrict__ hprev,  // [B, H] bf16  (own h at t-1)
    const unsigned short* __restrict__ wih,    // [4H, H] bf16
    const unsigned short* __restrict__ whh,    // [4H, H] bf16
    const float* __restrict__ bias,            // [4H] combined
    float* __restrict__ c_state,               // [B, H] fp32 in/out (1 lane per elem)
    unsigned short* __restrict__ hout_bf,      // [B, H] bf16 (own h at t)
    float* __restrict__ hout_f32)              // [B, H] fp32 to d_out, or nullptr
{
    __shared__ unsigned short s_a[32 * H_];    // 64 KB staged A rows

    const int mtile0 = (blockIdx.x >> 3) * 32;                     // 4 M-pairs
    const int ntile  = (((blockIdx.x & 7) << 3) + threadIdx.y) << 4;  // 64 n-tiles
    const int lane   = threadIdx.x;
    const int khalf  = lane >> 4;
    const int lm     = lane & 15;
    const int nr     = ntile + lm;

    v8f ai0 = {}, af0 = {}, ag0 = {}, ao0 = {};
    v8f ai1 = {}, af1 = {}, ag1 = {}, ao1 = {};

    // ---- phase 1: hin @ Wih^T ----
    stage_tile(s_a, hin + (size_t)mtile0 * H_);
    for (int k = 0; k < H_; k += 32) {
        v16bf a0 = load_frag_bf16(s_a, lm,      H_, k, khalf);
        v16bf a1 = load_frag_bf16(s_a, lm + 16, H_, k, khalf);
        v16bf b0 = load_frag_bf16(wih, nr,          H_, k, khalf);
        v16bf b1 = load_frag_bf16(wih, nr + H_,     H_, k, khalf);
        v16bf b2 = load_frag_bf16(wih, nr + 2 * H_, H_, k, khalf);
        v16bf b3 = load_frag_bf16(wih, nr + 3 * H_, H_, k, khalf);
        ai0 = WMMA_BF16(a0, b0, ai0);  ai1 = WMMA_BF16(a1, b0, ai1);
        af0 = WMMA_BF16(a0, b1, af0);  af1 = WMMA_BF16(a1, b1, af1);
        ag0 = WMMA_BF16(a0, b2, ag0);  ag1 = WMMA_BF16(a1, b2, ag1);
        ao0 = WMMA_BF16(a0, b3, ao0);  ao1 = WMMA_BF16(a1, b3, ao1);
    }
    __syncthreads();   // all waves done reading s_a before re-stage

    // ---- phase 2: hprev @ Whh^T ----
    stage_tile(s_a, hprev + (size_t)mtile0 * H_);
    for (int k = 0; k < H_; k += 32) {
        v16bf a0 = load_frag_bf16(s_a, lm,      H_, k, khalf);
        v16bf a1 = load_frag_bf16(s_a, lm + 16, H_, k, khalf);
        v16bf b0 = load_frag_bf16(whh, nr,          H_, k, khalf);
        v16bf b1 = load_frag_bf16(whh, nr + H_,     H_, k, khalf);
        v16bf b2 = load_frag_bf16(whh, nr + 2 * H_, H_, k, khalf);
        v16bf b3 = load_frag_bf16(whh, nr + 3 * H_, H_, k, khalf);
        ai0 = WMMA_BF16(a0, b0, ai0);  ai1 = WMMA_BF16(a1, b0, ai1);
        af0 = WMMA_BF16(a0, b1, af0);  af1 = WMMA_BF16(a1, b1, af1);
        ag0 = WMMA_BF16(a0, b2, ag0);  ag1 = WMMA_BF16(a1, b2, ag1);
        ao0 = WMMA_BF16(a0, b3, ao0);  ao1 = WMMA_BF16(a1, b3, ao1);
    }

    const float bi_ = bias[nr];
    const float bf_ = bias[nr + H_];
    const float bg_ = bias[nr + 2 * H_];
    const float bo_ = bias[nr + 3 * H_];
#pragma unroll
    for (int sub = 0; sub < 2; ++sub) {
        const int mbase = mtile0 + sub * 16;
        const v8f& vi = sub ? ai1 : ai0;
        const v8f& vf = sub ? af1 : af0;
        const v8f& vg = sub ? ag1 : ag0;
        const v8f& vo = sub ? ao1 : ao0;
#pragma unroll
        for (int v = 0; v < 8; ++v) {
            const int m = mbase + khalf * 8 + v;
            const size_t idx = (size_t)m * H_ + nr;
            float gi = vi[v] + bi_;
            float gf = vf[v] + bf_;
            float gg = vg[v] + bg_;
            float go = vo[v] + bo_;
            float cp = c_state[idx];
            float cn = sigmoidf_(gf) * cp + sigmoidf_(gi) * tanhf(gg);
            float hn = sigmoidf_(go) * tanhf(cn);
            c_state[idx] = cn;
            hout_bf[idx] = f32_to_bf16_rne(hn);
            if (hout_f32) hout_f32[idx] = hn;
        }
    }
}

// ---------------- host launcher ----------------

static inline int cdiv(int a, int b) { return (a + b - 1) / b; }

extern "C" void kernel_launch(void* const* d_in, const int* in_sizes, int n_in,
                              void* d_out, int out_size, void* d_ws, size_t ws_size,
                              hipStream_t stream) {
    (void)in_sizes; (void)n_in; (void)out_size; (void)ws_size;
    const float* x     = (const float*)d_in[0];  // [B, T, E]
    const float* Wih0  = (const float*)d_in[1];  // [4H, E]
    // d_in[2] = W_hh0: multiplies a zero h -> unused (bias still counts)
    const float* bih0  = (const float*)d_in[3];  // [4H]
    const float* bhh0  = (const float*)d_in[4];  // [4H]
    const float* Wihr  = (const float*)d_in[5];  // [2, 4H, H]
    const float* Whhr  = (const float*)d_in[6];  // [2, 4H, H]
    const float* bihr  = (const float*)d_in[7];  // [2, 4H]
    const float* bhhr  = (const float*)d_in[8];  // [2, 4H]
    float* out = (float*)d_out;                  // [T, B, H]

    // workspace carve-up (bf16 stored as ushort)
    char* ws = (char*)d_ws;
    size_t off = 0;
    unsigned short* x_bf    = (unsigned short*)(ws + off); off += (size_t)B_ * T_ * E_ * 2;     // 32 MB
    unsigned short* wih0_bf = (unsigned short*)(ws + off); off += (size_t)FH_ * E_ * 2;         //  4 MB
    unsigned short* wihr_bf = (unsigned short*)(ws + off); off += (size_t)2 * FH_ * H_ * 2;     // 16 MB
    unsigned short* whhr_bf = (unsigned short*)(ws + off); off += (size_t)2 * FH_ * H_ * 2;     // 16 MB
    float*          bias    = (float*)(ws + off);          off += (size_t)3 * FH_ * 4;
    unsigned short* h0_bf   = (unsigned short*)(ws + off); off += (size_t)T_ * B_ * H_ * 2;     // 64 MB
    unsigned short* hstate  = (unsigned short*)(ws + off); off += (size_t)2 * 2 * B_ * H_ * 2;  // [layer][parity][B*H]
    float*          cstate  = (float*)(ws + off);          off += (size_t)2 * B_ * H_ * 4;      // [layer][B*H]

    // 1) convert inputs/weights to bf16, combine biases, zero state (every call: deterministic)
    {
        int n;
        n = B_ * T_ * E_;
        cvt_f32_bf16_k<<<cdiv(n, 256), 256, 0, stream>>>(x, x_bf, n);
        n = FH_ * E_;
        cvt_f32_bf16_k<<<cdiv(n, 256), 256, 0, stream>>>(Wih0, wih0_bf, n);
        n = 2 * FH_ * H_;
        cvt_f32_bf16_k<<<cdiv(n, 256), 256, 0, stream>>>(Wihr, wihr_bf, n);
        cvt_f32_bf16_k<<<cdiv(n, 256), 256, 0, stream>>>(Whhr, whhr_bf, n);
        combine_bias_k<<<cdiv(FH_, 256), 256, 0, stream>>>(bih0, bhh0, bihr, bhhr, bias);
        int zwords = (int)(((size_t)2 * 2 * B_ * H_ * 2 + (size_t)2 * B_ * H_ * 4) / 4);
        zero_u32_k<<<cdiv(zwords, 256), 256, 0, stream>>>((unsigned int*)hstate, zwords);
    }

    // 2) layer 0 for all timesteps at once (state always zero -> time-parallel GEMM)
    {
        const int mtiles = (B_ * T_) / 16;  // 2048
        const int ntiles = H_ / 16;         // 64
        dim3 blk(32, 8);
        lstm_layer0_wmma<<<(mtiles * ntiles) / 8, blk, 0, stream>>>(x_bf, wih0_bf, bias, h0_bf);
    }

    // 3) recurrent layers 1..2, sequential over T; h double-buffered, c in place
    const size_t WSZ = (size_t)FH_ * H_;   // elems per weight matrix
    const size_t SSZ = (size_t)B_ * H_;    // elems per state plane
    dim3 blk(32, 8);
    const int cellBlocks = 32;             // 4 M-pairs x 8 n-blocks
    for (int t = 0; t < T_; ++t) {
        const int p0 = t & 1, p1 = p0 ^ 1;
        unsigned short* h1_prev = hstate + (0 * 2 + p0) * SSZ;
        unsigned short* h1_cur  = hstate + (0 * 2 + p1) * SSZ;
        unsigned short* h2_prev = hstate + (2 + p0) * SSZ;
        unsigned short* h2_cur  = hstate + (2 + p1) * SSZ;
        // layer 1: input = h0[t]
        lstm_cell_wmma<<<cellBlocks, blk, 0, stream>>>(
            h0_bf + (size_t)t * SSZ, h1_prev,
            wihr_bf, whhr_bf, bias + FH_,
            cstate, h1_cur, nullptr);
        // layer 2: input = layer-1 output at t; also writes fp32 d_out[t]
        lstm_cell_wmma<<<cellBlocks, blk, 0, stream>>>(
            h1_cur, h2_prev,
            wihr_bf + WSZ, whhr_bf + WSZ, bias + 2 * FH_,
            cstate + SSZ, h2_cur, out + (size_t)t * SSZ);
    }
}
